// QuadraticRNN_23124103922001
// MI455X (gfx1250) — matrix operand order repaired and checked by reading, not compiled, not run on hardware
//
#include <hip/hip_runtime.h>

// ---------------------------------------------------------------------------
// QuadraticRNN for MI455X (gfx1250): bf16 WMMA with f32 accumulation.
// Compute-bound (~203 GFLOP vs ~6us HBM); weights/h/x all L2-resident.
// Block = 1 N-group (64 cols) x 8 waves; each wave computes TWO 16-row
// M-tiles sharing one set of B fragments -> 1.0 ds_load per WMMA and every
// other WMMA issues with no dscnt wait. B panel staged in LDS double-buffered
// K chunks via global_load_async_to_lds_b128 + s_wait_asynccnt.
// ---------------------------------------------------------------------------

typedef __bf16 bf16_t;
typedef __attribute__((ext_vector_type(16))) __bf16 v16bf;
typedef __attribute__((ext_vector_type(8)))  __bf16 v8bf;
typedef __attribute__((ext_vector_type(8)))  float  v8f;
typedef __attribute__((ext_vector_type(4)))  float  v4f;

#define B_SZ 1024
#define K_SZ 64
#define P_SZ 512
#define D_SZ 1024
#define KC   128   // K-chunk staged in LDS (64 cols x 128 x bf16 = 16KB)

#if defined(__has_builtin)
#if __has_builtin(__builtin_amdgcn_global_load_async_to_lds_b128)
#define HAVE_ASYNC_LDS 1
#endif
#endif

#ifdef HAVE_ASYNC_LDS
// Builtin signature (from hipcc diagnostic): param0 is AS(1) pointer to a
// 4 x int vector, param1 is the LDS-side pointer, then imm offset, imm cpol.
typedef int v4i_ __attribute__((vector_size(16)));
typedef v4i_ __attribute__((address_space(1)))* as1_v4i_p;
typedef v4i_ __attribute__((address_space(3)))* as3_v4i_p;
#endif

__device__ __forceinline__ void wait_async() {
#ifdef HAVE_ASYNC_LDS
  asm volatile("s_wait_asynccnt 0" ::: "memory");
#endif
}

// ---- fragment loaders (ISA 7.12.2 layouts) --------------------------------

// A 16x32 bf16 from f32 row-major (row stride ld floats).
__device__ __forceinline__ v16bf fragA_f32(const float* __restrict__ base,
                                           int ld, int lane, int k0) {
  const int r  = lane & 15;
  const int kg = (lane >> 4) << 3;       // 0 or 8
  const float* p = base + (size_t)r * ld + k0 + kg;
  v4f f0 = *(const v4f*)(p);
  v4f f1 = *(const v4f*)(p + 4);
  v4f f2 = *(const v4f*)(p + 16);
  v4f f3 = *(const v4f*)(p + 20);
  v16bf a;
#pragma unroll
  for (int i = 0; i < 4; ++i) {
    a[i]      = (bf16_t)f0[i];
    a[i + 4]  = (bf16_t)f1[i];
    a[i + 8]  = (bf16_t)f2[i];
    a[i + 12] = (bf16_t)f3[i];
  }
  return a;
}

// A 16x32 bf16 from bf16 row-major.
__device__ __forceinline__ v16bf fragA_bf16(const bf16_t* __restrict__ base,
                                            int ld, int lane, int k0) {
  const int r  = lane & 15;
  const int kg = (lane >> 4) << 3;       // 0 or 8
  const bf16_t* p = base + (size_t)r * ld + k0 + kg;
  v8bf lo = *(const v8bf*)(p);
  v8bf hi = *(const v8bf*)(p + 16);
  v16bf a;
#pragma unroll
  for (int i = 0; i < 8; ++i) { a[i] = lo[i]; a[i + 8] = hi[i]; }
  return a;
}

// B 32x16 bf16 from LDS panel (row n = output column, row stride KC).
__device__ __forceinline__ v16bf fragB_lds(const bf16_t* base, int lane, int k0) {
  const int n  = lane & 15;
  const int kg = (lane >> 4) << 3;       // 0 or 8
  const bf16_t* p = base + n * KC + k0 + kg;
  v8bf lo = *(const v8bf*)(p);
  v8bf hi = *(const v8bf*)(p + 16);
  v16bf b;
#pragma unroll
  for (int i = 0; i < 8; ++i) { b[i] = lo[i]; b[i + 8] = hi[i]; }
  return b;
}

#define WMMA_BF16(A, Bm, C) \
  __builtin_amdgcn_wmma_f32_16x16x32_bf16(false, (A), false, (Bm), (short)0, (C), false, false)

// ---- B-panel staging: 64 rows x KC bf16 into LDS --------------------------
// 256 threads: 4 threads per row, 64B (32 bf16) each = 4x b128 per thread.

__device__ __forceinline__ void stage(const bf16_t* __restrict__ w, int ldw,
                                      bf16_t* lbuf, int k0, int tid) {
  const int row = tid >> 2;              // 0..63
  const int seg = (tid & 3) * 32;        // bf16 elements
  const bf16_t* g = w + (size_t)row * ldw + k0 + seg;
  bf16_t* l = lbuf + row * KC + seg;
#ifdef HAVE_ASYNC_LDS
#pragma unroll
  for (int i = 0; i < 4; ++i) {
    v4i_* gp = (v4i_*)(g + i * 8);       // generic, const stripped
    v4i_* lp = (v4i_*)(l + i * 8);       // generic
    __builtin_amdgcn_global_load_async_to_lds_b128(
        (as1_v4i_p)gp, (as3_v4i_p)lp, 0, 0);
  }
#else
  v8bf* lv = (v8bf*)l;
  const v8bf* gv = (const v8bf*)g;
  lv[0] = gv[0]; lv[1] = gv[1]; lv[2] = gv[2]; lv[3] = gv[3];
#endif
}

// ---- GEMM panel, 2 M-tiles per wave ---------------------------------------
// acc[m][0..3] += A_m(16 x Ktot) x W(64 cols x Ktot)^T, m = 0,1.
// Both M-tiles consume the same B fragment: 2 WMMAs per ds_load pair.

template <bool A_IS_F32>
__device__ __forceinline__ void gemm_panel2(v8f acc[2][4], const void* Asrc, int lda,
                                            const bf16_t* __restrict__ Wcols, int ldw,
                                            int Ktot, bf16_t (*sB)[64 * KC],
                                            int tid, int lane) {
  stage(Wcols, ldw, sB[0], 0, tid);
  wait_async();
  __syncthreads();
  const int NC = Ktot / KC;
  for (int c = 0; c < NC; ++c) {
    if (c + 1 < NC) stage(Wcols, ldw, sB[(c + 1) & 1], (c + 1) * KC, tid);
    const bf16_t* buf = sB[c & 1];
#pragma unroll
    for (int kk = 0; kk < KC; kk += 32) {
      v16bf a0, a1;
      if constexpr (A_IS_F32) {
        a0 = fragA_f32((const float*)Asrc, lda, lane, c * KC + kk);
        a1 = fragA_f32((const float*)Asrc + (size_t)16 * lda, lda, lane, c * KC + kk);
      } else {
        a0 = fragA_bf16((const bf16_t*)Asrc, lda, lane, c * KC + kk);
        a1 = fragA_bf16((const bf16_t*)Asrc + (size_t)16 * lda, lda, lane, c * KC + kk);
      }
#pragma unroll
      for (int j = 0; j < 4; ++j) {
        v16bf b = fragB_lds(buf + j * 16 * KC, lane, kk);
        acc[0][j] = WMMA_BF16(a0, b, acc[0][j]);
        acc[1][j] = WMMA_BF16(a1, b, acc[1][j]);
      }
    }
    wait_async();
    __syncthreads();
  }
}

// ---- epilogue: square + store bf16 (2 M-tiles) ----------------------------

__device__ __forceinline__ void store_sq_bf16(bf16_t* __restrict__ hout,
                                              const v8f acc[2][4], int m0, int ng, int lane) {
  const int n  = lane & 15;
  const int mo = (lane >> 4) << 3;
#pragma unroll
  for (int m = 0; m < 2; ++m)
#pragma unroll
    for (int j = 0; j < 4; ++j)
#pragma unroll
      for (int v = 0; v < 8; ++v) {
        float s = acc[m][j][v];
        hout[(size_t)(m0 + m * 16 + mo + v) * D_SZ + ng + j * 16 + n] = (bf16_t)(s * s);
      }
}

// ---- weight conversion ----------------------------------------------------

__global__ void k_convert(const float* __restrict__ src, bf16_t* __restrict__ dst, int n) {
  int i = (blockIdx.x * blockDim.x + threadIdx.x) * 4;
  if (i + 3 < n) {
    v4f v = *(const v4f*)(src + i);
    dst[i]     = (bf16_t)v[0];
    dst[i + 1] = (bf16_t)v[1];
    dst[i + 2] = (bf16_t)v[2];
    dst[i + 3] = (bf16_t)v[3];
  }
}

// ---- init: h = (x0 @ Win^T + x1 @ Wdrive^T)^2 -----------------------------
// grid: 16 N-groups x 4 M-chunks = 64 blocks, 8 waves, 2 M-tiles per wave.

__global__ void __launch_bounds__(256)
k_init(const float* __restrict__ x, const bf16_t* __restrict__ win,
       const bf16_t* __restrict__ wdr, bf16_t* __restrict__ hout) {
  __shared__ __align__(16) bf16_t sB[2][64 * KC];
  const int tid = threadIdx.x, lane = tid & 31, wv = tid >> 5;
  const int ng = (blockIdx.x & 15) * 64;
  const int m0 = ((blockIdx.x >> 4) * 8 + wv) * 32;   // 2 tiles of 16 rows

  const float* x0 = x + (size_t)m0 * K_SZ * P_SZ;     // t = 0
  const float* x1 = x0 + P_SZ;                         // t = 1

  v8f acc[2][4] = {};
  gemm_panel2<true>(acc, x0, K_SZ * P_SZ, win + (size_t)ng * P_SZ, P_SZ, P_SZ, sB, tid, lane);
  gemm_panel2<true>(acc, x1, K_SZ * P_SZ, wdr + (size_t)ng * P_SZ, P_SZ, P_SZ, sB, tid, lane);
  store_sq_bf16(hout, acc, m0, ng, lane);
}

// ---- one step: h' = (h @ Wmix^T + x[:,t+2,:] @ Wdrive^T)^2 ----------------

__global__ void __launch_bounds__(256)
k_step(const bf16_t* __restrict__ hin, const float* __restrict__ x, int t,
       const bf16_t* __restrict__ wmix, const bf16_t* __restrict__ wdr,
       bf16_t* __restrict__ hout) {
  __shared__ __align__(16) bf16_t sB[2][64 * KC];
  const int tid = threadIdx.x, lane = tid & 31, wv = tid >> 5;
  const int ng = (blockIdx.x & 15) * 64;
  const int m0 = ((blockIdx.x >> 4) * 8 + wv) * 32;

  v8f acc[2][4] = {};
  gemm_panel2<false>(acc, hin + (size_t)m0 * D_SZ, D_SZ,
                     wmix + (size_t)ng * D_SZ, D_SZ, D_SZ, sB, tid, lane);
  gemm_panel2<true>(acc, x + (size_t)m0 * K_SZ * P_SZ + (size_t)(t + 2) * P_SZ, K_SZ * P_SZ,
                    wdr + (size_t)ng * P_SZ, P_SZ, P_SZ, sB, tid, lane);
  store_sq_bf16(hout, acc, m0, ng, lane);
}

// ---- output: out = h @ Wout^T (f32) ---------------------------------------
// grid: 8 N-groups (P=512) x 4 M-chunks = 32 blocks.

__global__ void __launch_bounds__(256)
k_out(const bf16_t* __restrict__ h, const bf16_t* __restrict__ wout,
      float* __restrict__ out) {
  __shared__ __align__(16) bf16_t sB[2][64 * KC];
  const int tid = threadIdx.x, lane = tid & 31, wv = tid >> 5;
  const int ng = (blockIdx.x & 7) * 64;
  const int m0 = ((blockIdx.x >> 3) * 8 + wv) * 32;

  v8f acc[2][4] = {};
  gemm_panel2<false>(acc, h + (size_t)m0 * D_SZ, D_SZ,
                     wout + (size_t)ng * D_SZ, D_SZ, D_SZ, sB, tid, lane);

  const int n  = lane & 15;
  const int mo = (lane >> 4) << 3;
#pragma unroll
  for (int m = 0; m < 2; ++m)
#pragma unroll
    for (int j = 0; j < 4; ++j)
#pragma unroll
      for (int v = 0; v < 8; ++v)
        out[(size_t)(m0 + m * 16 + mo + v) * P_SZ + ng + j * 16 + n] = acc[m][j][v];
}

// ---------------------------------------------------------------------------

extern "C" void kernel_launch(void* const* d_in, const int* in_sizes, int n_in,
                              void* d_out, int out_size, void* d_ws, size_t ws_size,
                              hipStream_t stream) {
  const float* x    = (const float*)d_in[0];  // (B,K,P)
  const float* Win  = (const float*)d_in[1];  // (D,P)
  const float* Wmix = (const float*)d_in[2];  // (D,D)
  const float* Wdr  = (const float*)d_in[3];  // (D,P)
  const float* Wout = (const float*)d_in[4];  // (P,D)

  char* ws = (char*)d_ws;
  const size_t MB = 1024u * 1024u;
  bf16_t* wmix_b = (bf16_t*)(ws + 0 * MB);   // D*D bf16 = 2MB
  bf16_t* win_b  = (bf16_t*)(ws + 2 * MB);   // D*P bf16 = 1MB
  bf16_t* wdr_b  = (bf16_t*)(ws + 3 * MB);   // D*P bf16 = 1MB
  bf16_t* wout_b = (bf16_t*)(ws + 4 * MB);   // P*D bf16 = 1MB
  bf16_t* hbuf0  = (bf16_t*)(ws + 5 * MB);   // B*D bf16 = 2MB
  bf16_t* hbuf1  = (bf16_t*)(ws + 7 * MB);   // B*D bf16 = 2MB

  k_convert<<<(D_SZ * D_SZ) / (256 * 4), 256, 0, stream>>>(Wmix, wmix_b, D_SZ * D_SZ);
  k_convert<<<(D_SZ * P_SZ) / (256 * 4), 256, 0, stream>>>(Win,  win_b,  D_SZ * P_SZ);
  k_convert<<<(D_SZ * P_SZ) / (256 * 4), 256, 0, stream>>>(Wdr,  wdr_b,  D_SZ * P_SZ);
  k_convert<<<(P_SZ * D_SZ) / (256 * 4), 256, 0, stream>>>(Wout, wout_b, P_SZ * D_SZ);

  k_init<<<64, 256, 0, stream>>>(x, win_b, wdr_b, hbuf0);

  bf16_t* bufs[2] = { hbuf0, hbuf1 };
  for (int t = 0; t < K_SZ - 2; ++t) {
    k_step<<<64, 256, 0, stream>>>(bufs[t & 1], x, t, wmix_b, wdr_b, bufs[(t + 1) & 1]);
  }
  k_out<<<32, 256, 0, stream>>>(bufs[(K_SZ - 2) & 1], wout_b, (float*)d_out);
}
